// WorldModel_60636348285562
// MI455X (gfx1250) — matrix operand order repaired and checked by reading, not compiled, not run on hardware
//
#include <hip/hip_runtime.h>
#include <hip/hip_bf16.h>
#include <cstdint>
#include <cstddef>

// ---------------------------------------------------------------------------
// Mamba (4 layers) forward for MI455X / gfx1250.
// GEMMs: v_wmma_f32_16x16x32_bf16, 256x128x32 block tile, 64x64 per wave,
// double-buffered LDS fed by GLOBAL_LOAD_ASYNC_TO_LDS_B128 (ASYNCcnt) when
// the toolchain exposes the builtins (guarded by __has_builtin).
// Scan keeps h-state in VGPRs; B/C broadcast through LDS.
// ---------------------------------------------------------------------------

typedef __attribute__((ext_vector_type(16))) __bf16 v16bf;
typedef __attribute__((ext_vector_type(8)))  __bf16 v8bf;
typedef __attribute__((ext_vector_type(8)))  float  v8f;
typedef __attribute__((__vector_size__(4 * sizeof(int)))) int v4i;   // matches builtin param

constexpr int LAYERS = 4;
constexpr int DM     = 512;    // d_model
constexpr int DI     = 1024;   // d_inner
constexpr int NS     = 64;     // d_state
constexpr int BATCH  = 256;
constexpr int SQ     = 117;
constexpr int BS     = BATCH * SQ;   // 29952 = 117 * 256

#if defined(__HIP_DEVICE_COMPILE__) && \
    __has_builtin(__builtin_amdgcn_global_load_async_to_lds_b128) && \
    __has_builtin(__builtin_amdgcn_s_wait_asynccnt)
#define HAVE_ASYNC_LDS 1
#else
#define HAVE_ASYNC_LDS 0
#endif

// ------------------------------- converters --------------------------------

__global__ void k_f32_to_bf16(const float* __restrict__ in, __bf16* __restrict__ out, int n) {
    int i = blockIdx.x * blockDim.x + threadIdx.x;
    if (i < n) out[i] = (__bf16)in[i];
}

// in: [K,N] f32 row-major  ->  out: [N,K] bf16 row-major (transposed)
__global__ void k_transpose_bf16(const float* __restrict__ in, __bf16* __restrict__ out,
                                 int K, int N) {
    int i = blockIdx.x * blockDim.x + threadIdx.x;
    if (i >= K * N) return;
    int k = i / N, n = i - k * N;
    out[(size_t)n * K + k] = (__bf16)in[i];
}

// ------------------------------- WMMA GEMM ---------------------------------
// C[M,N] = A[M,K](bf16) * Wt[N,K](bf16)^T, f32 accum.
// Block tile 256x128x32, 8 waves (wave32) arranged 4x2, each wave 64x64
// (4x4 WMMA frags). LDS is double-buffered; next K-tile streams in via
// async-to-LDS while the current tile is multiplied.

constexpr int BMT = 256, BNT = 128, BKT = 32;
constexpr int LDS_STRIDE = 40;   // halves per row; padded vs bank conflicts

__device__ inline void copy_b128_to_lds(const __bf16* g, __bf16* l) {
#if HAVE_ASYNC_LDS
    __builtin_amdgcn_global_load_async_to_lds_b128(
        (__attribute__((address_space(1))) v4i*)g,
        (__attribute__((address_space(3))) v4i*)l,
        0, 0);
#else
    *(v8bf*)l = *(const v8bf*)g;
#endif
}

__device__ inline void drain_lds_copies() {
#if HAVE_ASYNC_LDS
    __builtin_amdgcn_s_wait_asynccnt(0);
#endif
}

// A fragment (16x32 bf16): lane<16 -> M=lane,    K = {kb..kb+7, kb+16..kb+23}, kb=0
//                          lane>=16 -> M=lane-16, same with kb=8
__device__ inline v16bf ldfragA(const __bf16* s, int lane) {
    int kb = (lane < 16) ? 0 : 8;
    v8bf lo = *(const v8bf*)(s + kb);
    v8bf hi = *(const v8bf*)(s + kb + 16);
    v16bf r;
#pragma unroll
    for (int i = 0; i < 8; ++i) { r[i] = lo[i]; r[i + 8] = hi[i]; }
    return r;
}

// B fragment (32x16 bf16, stored [N][K] in LDS): lane<16 -> N=lane,  K = {kb..kb+7, kb+16..kb+23}, kb=0
//                                                lane>=16 -> N=lane-16, same with kb=8
__device__ inline v16bf ldfragB(const __bf16* s, int lane) {
    int kb = (lane < 16) ? 0 : 8;
    v8bf lo = *(const v8bf*)(s + kb);
    v8bf hi = *(const v8bf*)(s + kb + 16);
    v16bf r;
#pragma unroll
    for (int i = 0; i < 8; ++i) { r[i] = lo[i]; r[i + 8] = hi[i]; }
    return r;
}

// mode 0: out0[M,N] = v
// mode 1: split columns at N/2: out0 gets cols [0,N/2), out1 gets [N/2,N)
// mode 2: out0[M,N] = softplus(v + bias[col])
__global__ __launch_bounds__(256)
void wmma_gemm_bf16(const __bf16* __restrict__ A, const __bf16* __restrict__ Wt,
                    int M, int N, int K, int mode,
                    float* __restrict__ out0, float* __restrict__ out1,
                    const float* __restrict__ bias) {
    __shared__ __bf16 As[2][BMT * LDS_STRIDE];   // 2 * 20 KB
    __shared__ __bf16 Bs[2][BNT * LDS_STRIDE];   // 2 * 10 KB

    const int m0   = blockIdx.y * BMT;
    const int n0   = blockIdx.x * BNT;
    const int tid  = threadIdx.x;
    const int lane = tid & 31;
    const int wave = tid >> 5;          // 0..7
    const int wm   = wave >> 1;         // 0..3  (64-row strips)
    const int wn   = wave & 1;          // 0..1  (64-col strips)

    v8f zero = {0.f, 0.f, 0.f, 0.f, 0.f, 0.f, 0.f, 0.f};
    v8f acc[4][4];
#pragma unroll
    for (int i = 0; i < 4; ++i)
#pragma unroll
        for (int j = 0; j < 4; ++j) acc[i][j] = zero;

    // global->LDS mapping:
    //   A tile 256x32: thread t covers row t, 4 x b128 chunks (32 halves)
    //   B tile 128x32: thread t covers row t>>1, segment (t&1)*16, 2 x b128
    const __bf16* gArow = A  + (size_t)(m0 + tid) * K;
    const __bf16* gBrow = Wt + (size_t)(n0 + (tid >> 1)) * K + (tid & 1) * 16;
    const int sArow = tid * LDS_STRIDE;
    const int sBrow = (tid >> 1) * LDS_STRIDE + (tid & 1) * 16;

    auto stage_tile = [&](int k, int buf) {
#pragma unroll
        for (int c = 0; c < 4; ++c)
            copy_b128_to_lds(gArow + k + c * 8, &As[buf][sArow + c * 8]);
        copy_b128_to_lds(gBrow + k,     &Bs[buf][sBrow]);
        copy_b128_to_lds(gBrow + k + 8, &Bs[buf][sBrow + 8]);
    };

    // prologue: stage tile 0
    stage_tile(0, 0);
    drain_lds_copies();
    __syncthreads();

    const int nk = K / BKT;
    for (int kt = 0; kt < nk; ++kt) {
        const int cur = kt & 1;
        if (kt + 1 < nk) {
            stage_tile((kt + 1) * BKT, cur ^ 1);           // overlap with WMMA
            if (kt + 2 < nk) {                             // warm L2 for t+2
                __builtin_prefetch(gArow + (kt + 2) * BKT, 0, 1);
                __builtin_prefetch(gBrow + (kt + 2) * BKT, 0, 1);
            }
        }

        v16bf af[4], bfm[4];
#pragma unroll
        for (int i = 0; i < 4; ++i)
            af[i] = ldfragA(&As[cur][(wm * 64 + i * 16 + (lane & 15)) * LDS_STRIDE], lane);
#pragma unroll
        for (int j = 0; j < 4; ++j)
            bfm[j] = ldfragB(&Bs[cur][(wn * 64 + j * 16 + (lane & 15)) * LDS_STRIDE], lane);
#pragma unroll
        for (int i = 0; i < 4; ++i)
#pragma unroll
            for (int j = 0; j < 4; ++j)
                acc[i][j] = __builtin_amdgcn_wmma_f32_16x16x32_bf16(
                    false, af[i], false, bfm[j], (short)0, acc[i][j], false, false);

        drain_lds_copies();    // next tile fully in LDS
        __syncthreads();       // all waves done reading 'cur'
    }

    // C/D layout: lanes 0-15 -> N=lane, M=r;  lanes 16-31 -> N=lane-16, M=r+8
    const int rbase = (lane < 16) ? 0 : 8;
    const int ncl   = lane & 15;
    const int nh    = N >> 1;
#pragma unroll
    for (int i = 0; i < 4; ++i) {
#pragma unroll
        for (int j = 0; j < 4; ++j) {
            const int row0 = m0 + wm * 64 + i * 16 + rbase;
            const int col  = n0 + wn * 64 + j * 16 + ncl;
#pragma unroll
            for (int r = 0; r < 8; ++r) {
                const size_t mrow = (size_t)(row0 + r);
                float v = acc[i][j][r];
                if (mode == 0) {
                    out0[mrow * N + col] = v;
                } else if (mode == 1) {
                    if (col < nh) out0[mrow * nh + col] = v;
                    else          out1[mrow * nh + (col - nh)] = v;
                } else { // softplus(v + bias)
                    float x = v + bias[col];
                    float sp = (x > 20.f) ? x : __logf(1.f + __expf(x));
                    out0[mrow * N + col] = sp;
                }
            }
        }
    }
}

// --------------------------- depthwise conv + SiLU --------------------------

__global__ void k_conv_silu(const float* __restrict__ xm, const float* __restrict__ cw,
                            const float* __restrict__ cb, __bf16* __restrict__ xc) {
    int t = blockIdx.x * blockDim.x + threadIdx.x;   // over BATCH*DI
    int d = t & (DI - 1);
    int b = t >> 10;
    float w0 = cw[d * 4 + 0], w1 = cw[d * 4 + 1], w2 = cw[d * 4 + 2], w3 = cw[d * 4 + 3];
    float bias = cb[d];
    float x0 = 0.f, x1 = 0.f, x2 = 0.f;               // left pad (K-1 = 3)
    for (int s = 0; s < SQ; ++s) {
        size_t idx = ((size_t)(b * SQ + s)) * DI + d;
        float xs = xm[idx];
        float a  = bias + w0 * x0 + w1 * x1 + w2 * x2 + w3 * xs;
        float sg = 1.f / (1.f + __expf(-a));
        xc[idx]  = (__bf16)(a * sg);
        x0 = x1; x1 = x2; x2 = xs;
    }
}

// ------------------------------ selective scan ------------------------------
// grid (DI/256, BATCH), block 256. One thread per (b,d); h[64]+A[64] in VGPRs,
// B/C per timestep broadcast via LDS. y written in-place over dt.

__global__ __launch_bounds__(256)
void k_scan(const __bf16* __restrict__ xc, float* __restrict__ dty,
            const float* __restrict__ bp, const float* __restrict__ cp,
            const float* __restrict__ alog, const float* __restrict__ Dp) {
    const int d = blockIdx.x * 256 + threadIdx.x;
    const int b = blockIdx.y;
    float A[NS], h[NS];
#pragma unroll
    for (int n = 0; n < NS; ++n) {
        A[n] = -__expf(alog[(size_t)d * NS + n]);
        h[n] = 0.f;
    }
    const float Dd = Dp[d];
    __shared__ float Bsh[NS];
    __shared__ float Csh[NS];
    for (int s = 0; s < SQ; ++s) {
        const int base = b * SQ + s;
        if (threadIdx.x < NS)            Bsh[threadIdx.x]      = bp[(size_t)base * NS + threadIdx.x];
        else if (threadIdx.x < 2 * NS)   Csh[threadIdx.x - NS] = cp[(size_t)base * NS + threadIdx.x - NS];
        __syncthreads();
        const size_t idx = (size_t)base * DI + d;
        const float dtt = dty[idx];
        const float xt  = (float)xc[idx];
        const float dx  = dtt * xt;
        float y = Dd * xt;
#pragma unroll
        for (int n = 0; n < NS; ++n) {
            h[n] = __expf(dtt * A[n]) * h[n] + dx * Bsh[n];
            y   += h[n] * Csh[n];
        }
        dty[idx] = y;
        __syncthreads();
    }
}

// ------------------------------- gate: y*silu(z) ----------------------------

__global__ void k_gate(const float* __restrict__ y, const float* __restrict__ z,
                       __bf16* __restrict__ ybf, int n) {
    int i = blockIdx.x * blockDim.x + threadIdx.x;
    if (i >= n) return;
    float zz = z[i];
    float sg = 1.f / (1.f + __expf(-zz));
    ybf[i] = (__bf16)(y[i] * zz * sg);
}

// --------------------------- residual + layernorm ---------------------------
// one block per row (512 cols, 128 threads x 4)

__global__ __launch_bounds__(128)
void k_res_ln(const float* __restrict__ proj, const float* __restrict__ xin,
              const float* __restrict__ g, const float* __restrict__ bta,
              float* __restrict__ xout) {
    const int row = blockIdx.x;
    const int t   = threadIdx.x;
    float v[4];
    float s1 = 0.f, s2 = 0.f;
#pragma unroll
    for (int i = 0; i < 4; ++i) {
        int c = t + i * 128;
        v[i] = proj[(size_t)row * DM + c] + xin[(size_t)row * DM + c];
        s1 += v[i];
        s2 += v[i] * v[i];
    }
    __shared__ float r1[128], r2[128];
    r1[t] = s1; r2[t] = s2;
    __syncthreads();
    for (int off = 64; off > 0; off >>= 1) {
        if (t < off) { r1[t] += r1[t + off]; r2[t] += r2[t + off]; }
        __syncthreads();
    }
    const float mu  = r1[0] * (1.f / DM);
    const float var = r2[0] * (1.f / DM) - mu * mu;
    const float rs  = rsqrtf(var + 1e-5f);
#pragma unroll
    for (int i = 0; i < 4; ++i) {
        int c = t + i * 128;
        xout[(size_t)row * DM + c] = (v[i] - mu) * rs * g[c] + bta[c];
    }
}

// --------------------------------- launch -----------------------------------

extern "C" void kernel_launch(void* const* d_in, const int* in_sizes, int n_in,
                              void* d_out, int out_size, void* d_ws, size_t ws_size,
                              hipStream_t stream) {
    const float* d_x     = (const float*)d_in[0];
    const float* d_inw   = (const float*)d_in[1];   // [L,512,2048]
    const float* d_convw = (const float*)d_in[2];   // [L,1024,1,4]
    const float* d_convb = (const float*)d_in[3];   // [L,1024]
    const float* d_xpw   = (const float*)d_in[4];   // [L,1024,128]
    const float* d_dtw   = (const float*)d_in[5];   // [L,1024,1024]
    const float* d_dtb   = (const float*)d_in[6];   // [L,1024]
    const float* d_alog  = (const float*)d_in[7];   // [L,1024,64]
    const float* d_Dp    = (const float*)d_in[8];   // [L,1024]
    const float* d_outw  = (const float*)d_in[9];   // [L,1024,512]
    const float* d_lng   = (const float*)d_in[10];  // [L,512]
    const float* d_lnb   = (const float*)d_in[11];  // [L,512]
    float* outp = (float*)d_out;

    char* ws = (char*)d_ws;
    size_t off = 0;
    auto take = [&](size_t bytes) -> char* {
        off = (off + 255) & ~(size_t)255;
        char* p = ws + off;
        off += bytes;
        return p;
    };
    __bf16* wT_in  = (__bf16*)take((size_t)LAYERS * 2048 * DM * 2);
    __bf16* wT_xp  = (__bf16*)take((size_t)LAYERS * 128  * DI * 2);
    __bf16* wT_dt  = (__bf16*)take((size_t)LAYERS * DI   * DI * 2);
    __bf16* wT_out = (__bf16*)take((size_t)LAYERS * DM   * DI * 2);
    __bf16* abuf   = (__bf16*)take((size_t)BS * DM * 2);
    float*  xm     = (float*) take((size_t)BS * DI * 4);
    float*  zb     = (float*) take((size_t)BS * DI * 4);
    __bf16* xcbf   = (__bf16*)take((size_t)BS * DI * 2);
    float*  dtb    = (float*) take((size_t)BS * DI * 4);   // dt, then y in-place
    float*  bpb    = (float*) take((size_t)BS * NS * 4);
    float*  cpb    = (float*) take((size_t)BS * NS * 4);
    __bf16* ybf    = (__bf16*)take((size_t)BS * DI * 2);
    float*  projb  = (float*) take((size_t)BS * DM * 4);
    float*  xcur   = (float*) take((size_t)BS * DM * 4);

    // ---- weight conversion + transpose (bf16, [N,K]) ----
    for (int l = 0; l < LAYERS; ++l) {
        {
            int n = DM * 2048;
            k_transpose_bf16<<<(n + 255) / 256, 256, 0, stream>>>(
                d_inw + (size_t)l * n, wT_in + (size_t)l * n, DM, 2048);
        }
        {
            int n = DI * 128;
            k_transpose_bf16<<<(n + 255) / 256, 256, 0, stream>>>(
                d_xpw + (size_t)l * n, wT_xp + (size_t)l * n, DI, 128);
        }
        {
            int n = DI * DI;
            k_transpose_bf16<<<(n + 255) / 256, 256, 0, stream>>>(
                d_dtw + (size_t)l * n, wT_dt + (size_t)l * n, DI, DI);
        }
        {
            int n = DI * DM;
            k_transpose_bf16<<<(n + 255) / 256, 256, 0, stream>>>(
                d_outw + (size_t)l * n, wT_out + (size_t)l * n, DI, DM);
        }
    }

    const int mtiles = BS / BMT;   // 117
    for (int l = 0; l < LAYERS; ++l) {
        const float* x_in  = (l == 0) ? d_x : xcur;
        float*       x_out = (l == LAYERS - 1) ? outp : xcur;

        // 1) x -> bf16
        {
            int n = BS * DM;
            k_f32_to_bf16<<<(n + 255) / 256, 256, 0, stream>>>(x_in, abuf, n);
        }
        // 2) in_proj: [BS,512] x [512,2048] -> xm | z
        wmma_gemm_bf16<<<dim3(2048 / BNT, mtiles), 256, 0, stream>>>(
            abuf, wT_in + (size_t)l * 2048 * DM, BS, 2048, DM, 1, xm, zb, nullptr);
        // 3) depthwise causal conv + SiLU -> xc (bf16)
        k_conv_silu<<<(BATCH * DI) / 256, 256, 0, stream>>>(
            xm, d_convw + (size_t)l * DI * 4, d_convb + (size_t)l * DI, xcbf);
        // 4) x_proj: [BS,1024] x [1024,128] -> B | C
        wmma_gemm_bf16<<<dim3(128 / BNT, mtiles), 256, 0, stream>>>(
            xcbf, wT_xp + (size_t)l * 128 * DI, BS, 128, DI, 1, bpb, cpb, nullptr);
        // 5) dt_proj + softplus: [BS,1024] x [1024,1024] -> dt
        wmma_gemm_bf16<<<dim3(DI / BNT, mtiles), 256, 0, stream>>>(
            xcbf, wT_dt + (size_t)l * DI * DI, BS, DI, DI, 2, dtb, nullptr,
            d_dtb + (size_t)l * DI);
        // 6) selective scan (y overwrites dt)
        k_scan<<<dim3(DI / 256, BATCH), 256, 0, stream>>>(
            xcbf, dtb, bpb, cpb, d_alog + (size_t)l * DI * NS, d_Dp + (size_t)l * DI);
        // 7) gate: y * silu(z) -> bf16
        {
            int n = BS * DI;
            k_gate<<<(n + 255) / 256, 256, 0, stream>>>(dtb, zb, ybf, n);
        }
        // 8) out_proj: [BS,1024] x [1024,512] -> proj
        wmma_gemm_bf16<<<dim3(DM / BNT, mtiles), 256, 0, stream>>>(
            ybf, wT_out + (size_t)l * DM * DI, BS, DM, DI, 0, projb, nullptr, nullptr);
        // 9) residual + layernorm
        k_res_ln<<<BS, 128, 0, stream>>>(
            projb, x_in, d_lng + (size_t)l * DM, d_lnb + (size_t)l * DM, x_out);
    }
}
